// DescripNet_35407710388659
// MI455X (gfx1250) — compile-verified, matrix operand order checked by reading, not run on hardware
//
#include <hip/hip_runtime.h>
#include <math.h>

// ---------------- problem constants (match reference) ----------------
#define BB   4
#define NN   4096
#define KNN  20
#define D0   64
#define D1   256
#define DF   512
#define DOUT 256
#define EPSBN 1e-5f
#define SLOPE 0.2f

typedef _Float16 v16h __attribute__((ext_vector_type(16)));
typedef float    v8f  __attribute__((ext_vector_type(8)));
typedef int      v4i  __attribute__((vector_size(16)));   // matches builtin's pointee

// ---------------- CDNA5 async global->LDS staging (guarded) ----------------
#if defined(__HIP_DEVICE_COMPILE__) && defined(__has_builtin)
# if __has_builtin(__builtin_amdgcn_global_load_async_to_lds_b128)
#  define USE_ASYNC_LDS 1
# endif
#endif

__device__ __forceinline__ void copy16_to_lds(const _Float16* g, _Float16* l) {
#ifdef USE_ASYNC_LDS
  __builtin_amdgcn_global_load_async_to_lds_b128(
      (__attribute__((address_space(1))) v4i*)g,
      (__attribute__((address_space(3))) v4i*)l, 0, 0);
#else
  *(uint4*)l = *(const uint4*)g;
#endif
}

__device__ __forceinline__ void wait_async_lds() {
#ifdef USE_ASYNC_LDS
# if __has_builtin(__builtin_amdgcn_s_wait_asynccnt)
  __builtin_amdgcn_s_wait_asynccnt(0);
# else
  asm volatile("s_wait_asynccnt 0x0" ::: "memory");
# endif
#endif
}

// ---------------- WMMA helpers (CDNA5 16x16x32 f16) ----------------
__device__ __forceinline__ v8f wmma_f16(v16h a, v16h b, v8f c) {
  return __builtin_amdgcn_wmma_f32_16x16x32_f16(false, a, false, b, (short)0, c, false, false);
}

// Load a 16x32 f16 fragment from LDS laid out row-major [16][stride] at K tile kt.
__device__ __forceinline__ v16h load_frag16(const _Float16* base, int stride, int lane, int kt) {
  int m  = lane & 15;
  int hi = lane >> 4;
  const _Float16* p = base + m * stride + kt * 32 + hi * 8;
  v16h f;
#pragma unroll
  for (int e = 0; e < 8; ++e) f[e] = p[e];
#pragma unroll
  for (int e = 0; e < 8; ++e) f[8 + e] = p[16 + e];
  return f;
}

// ---------------- register-resident top-20 (smallest) ----------------
struct TopK {
  float bd[KNN];
  int   bi[KNN];
  float worst;
  __device__ __forceinline__ void init() {
#pragma unroll
    for (int t = 0; t < KNN; ++t) { bd[t] = 3.4e38f; bi[t] = 0; }
    worst = 3.4e38f;
  }
  __device__ __forceinline__ void insert(float d, int i) {
    if (d < worst) {
      float w = -3.4e38f; int wi = 0;
#pragma unroll
      for (int t = 0; t < KNN; ++t) { if (bd[t] > w) { w = bd[t]; wi = t; } }
#pragma unroll
      for (int t = 0; t < KNN; ++t) { if (t == wi) { bd[t] = d; bi[t] = i; } }
      float nw = -3.4e38f;
#pragma unroll
      for (int t = 0; t < KNN; ++t) nw = fmaxf(nw, bd[t]);
      worst = nw;
    }
  }
};

// ---------------- small utility kernels ----------------
__global__ void zero_stats_kernel(float* s0, float* s1) {
  int t = blockIdx.x * blockDim.x + threadIdx.x;
  if (t < 2 * D0) s0[t] = 0.f;
  if (t < 2 * D1) s1[t] = 0.f;
}

__global__ void cvt_f16_kernel(const float* src, _Float16* dst, int n) {
  int t = blockIdx.x * blockDim.x + threadIdx.x;
  if (t < n) dst[t] = (_Float16)src[t];
}

__global__ void cvt_diff_f16_kernel(const float* a, const float* b, _Float16* dst, int n) {
  int t = blockIdx.x * blockDim.x + threadIdx.x;
  if (t < n) dst[t] = (_Float16)(a[t] - b[t]);
}

// ---------------- knn layer 0: D=3, brute force per point ----------------
__global__ __launch_bounds__(256) void knn0_kernel(const float* __restrict__ x, int* __restrict__ idx0) {
  int t = blockIdx.x * blockDim.x + threadIdx.x;
  if (t >= BB * NN) return;
  int b = t / NN, n = t % NN;
  const float* xb = x + (size_t)b * NN * 3;
  float xi0 = xb[n * 3 + 0], xi1 = xb[n * 3 + 1], xi2 = xb[n * 3 + 2];
  TopK tk; tk.init();
  for (int m = 0; m < NN; ++m) {
    float a = xb[m * 3 + 0] - xi0;
    float c = xb[m * 3 + 1] - xi1;
    float e = xb[m * 3 + 2] - xi2;
    tk.insert(a * a + c * c + e * e, m);
  }
#pragma unroll
  for (int j = 0; j < KNN; ++j) idx0[(size_t)t * KNN + j] = tk.bi[j];
}

// ---------------- layer-0 node projections: P0 = xWt0, Q0 = x(Wp0-Wt0) ----------------
__global__ void proj0_kernel(const float* __restrict__ x, const float* __restrict__ Wt0,
                             const float* __restrict__ Wp0, float* __restrict__ P0,
                             float* __restrict__ Q0) {
  int t = blockIdx.x * blockDim.x + threadIdx.x;
  if (t >= BB * NN * D0) return;
  int i = t >> 6, c = t & 63;
  float p = 0.f, q = 0.f;
#pragma unroll
  for (int d = 0; d < 3; ++d) {
    float xv = x[(size_t)i * 3 + d];
    float wt = Wt0[d * D0 + c];
    p += xv * wt;
    q += xv * (Wp0[d * D0 + c] - wt);
  }
  P0[t] = p; Q0[t] = q;
}

// ---------------- EdgeConv reduce: max over k + BN stats (generic over C) ----------------
template <int C, int PTS>
__global__ void edge_reduce_kernel(const float* __restrict__ P, const float* __restrict__ Q,
                                   const float* __restrict__ bt, const float* __restrict__ bp,
                                   const int* __restrict__ idx, float* __restrict__ hmax,
                                   float* __restrict__ stats) {
  int c = threadIdx.x;                       // channel
  int base = blockIdx.x * PTS;               // first point of this block
  float bias = bt[c] + bp[c];
  float bsum = 0.f, bsq = 0.f;
  for (int p = 0; p < PTS; ++p) {
    int i = base + p;
    int b = i / NN;
    float Pi = P[(size_t)i * C + c];
    const int* nb = idx + (size_t)i * KNN;
    float mx = -3.4e38f;
#pragma unroll 4
    for (int kk = 0; kk < KNN; ++kk) {
      int j = nb[kk];
      float m = Pi + Q[(size_t)(b * NN + j) * C + c] + bias;
      mx = fmaxf(mx, m);
      bsum += m; bsq += m * m;
    }
    hmax[(size_t)i * C + c] = mx;
  }
  atomicAdd(&stats[c], bsum);
  atomicAdd(&stats[C + c], bsq);
}

// ---------------- BN (post-max, valid for gamma>0) + leaky relu + f16 copy ----------------
template <int C>
__global__ void bn_leaky_kernel(const float* __restrict__ hmax, const float* __restrict__ stats,
                                const float* __restrict__ g, const float* __restrict__ be,
                                float* __restrict__ hout, _Float16* __restrict__ houth) {
  int t = blockIdx.x * blockDim.x + threadIdx.x;
  if (t >= BB * NN * C) return;
  int c = t & (C - 1);
  float cnt = (float)(BB * NN * KNN);
  float mu = stats[c] / cnt;
  float var = stats[C + c] / cnt - mu * mu;
  float v = (hmax[t] - mu) * rsqrtf(var + EPSBN) * g[c] + be[c];
  v = v > 0.f ? v : SLOPE * v;
  hout[t] = v;
  houth[t] = (_Float16)v;
}

// ---------------- squared norms for knn1 ----------------
__global__ void d2_kernel(const float* __restrict__ h0, float* __restrict__ d2) {
  int t = blockIdx.x * blockDim.x + threadIdx.x;
  if (t >= BB * NN) return;
  float s = 0.f;
#pragma unroll 8
  for (int c = 0; c < D0; ++c) { float v = h0[(size_t)t * D0 + c]; s += v * v; }
  d2[t] = s;
}

// ---------------- fused WMMA distance + streaming top-20 (knn layer 1) ----------------
// One wave (32 lanes) owns 32 rows; streams 16-column tiles of h·hT via WMMA.
// B tiles are double-buffered and staged with async global->LDS copies.
__global__ __launch_bounds__(32) void knn1_kernel(const _Float16* __restrict__ h0h,
                                                  const float* __restrict__ d2,
                                                  int* __restrict__ idx1) {
  __shared__ __align__(16) _Float16 Ar[32 * D0];
  __shared__ __align__(16) _Float16 Bc[2][16 * D0];
  __shared__ float DT[32][17];
  __shared__ float d2r[32];

  int lane = threadIdx.x;
  int blk = blockIdx.x;                  // BB * (NN/32)
  int b = blk / (NN / 32);
  int r0 = (blk % (NN / 32)) * 32;
  const _Float16* hb = h0h + (size_t)b * NN * D0;

  { // stage 32 A rows (each lane: its own row, 128B)
    const uint4* src = (const uint4*)(hb + (size_t)(r0 + lane) * D0);
    uint4* dst = (uint4*)(Ar + lane * D0);
#pragma unroll
    for (int i = 0; i < 8; ++i) dst[i] = src[i];
    d2r[lane] = d2[b * NN + r0 + lane];
  }
  __syncthreads();

  v16h a00 = load_frag16(Ar,           D0, lane, 0);   // rows 0-15,  K 0-31
  v16h a01 = load_frag16(Ar,           D0, lane, 1);   // rows 0-15,  K 32-63
  v16h a10 = load_frag16(Ar + 16 * D0, D0, lane, 0);   // rows 16-31, K 0-31
  v16h a11 = load_frag16(Ar + 16 * D0, D0, lane, 1);

  // stage a 16-column tile (2 KB): 2 lanes per column, 4x b128 each
  auto stage_tile = [&](int ct, int buf) {
    int col = lane >> 1, part = lane & 1;
    const _Float16* src = hb + (size_t)(ct * 16 + col) * D0 + part * 32;
    _Float16* dst = &Bc[buf][col * D0 + part * 32];
#pragma unroll
    for (int i = 0; i < 4; ++i) copy16_to_lds(src + i * 8, dst + i * 8);
  };

  TopK tk; tk.init();
  int n = lane & 15, hi = lane >> 4;

  stage_tile(0, 0);                       // prologue prefetch
  for (int ct = 0; ct < NN / 16; ++ct) {
    int cur = ct & 1;
    wait_async_lds();                     // tile ct landed in LDS
    __syncthreads();
    if (ct + 1 < NN / 16) stage_tile(ct + 1, cur ^ 1);   // overlap next fetch

    int c0 = ct * 16;
    v16h b0 = load_frag16(&Bc[cur][0], D0, lane, 0);
    v16h b1 = load_frag16(&Bc[cur][0], D0, lane, 1);
    v8f c0v = {}; c0v = wmma_f16(a00, b0, c0v); c0v = wmma_f16(a01, b1, c0v);
    v8f c1v = {}; c1v = wmma_f16(a10, b0, c1v); c1v = wmma_f16(a11, b1, c1v);

    float d2c = d2[b * NN + c0 + n];
#pragma unroll
    for (int v = 0; v < 8; ++v) {
      DT[hi * 8 + v][n]      = d2r[hi * 8 + v]      + d2c - 2.0f * c0v[v];
      DT[16 + hi * 8 + v][n] = d2r[16 + hi * 8 + v] + d2c - 2.0f * c1v[v];
    }
    __syncthreads();
#pragma unroll 4
    for (int j = 0; j < 16; ++j) tk.insert(DT[lane][j], c0 + j);
    __syncthreads();
  }
#pragma unroll
  for (int j = 0; j < KNN; ++j)
    idx1[((size_t)(b * NN + r0 + lane)) * KNN + j] = tk.bi[j];
}

// ---------------- WMMA f16 GEMM: C[M,Nc] = A[M,KD] * Bm[KD,Nc] ----------------
// 128-thread block = 4 waves computing a 16x64 output strip; A tile shared,
// K loop fully unrolled at compile time. Optional bias + relu on store.
template <int KD>
__global__ __launch_bounds__(128) void gemm_wmma_kernel(const _Float16* __restrict__ A,
                                                        const _Float16* __restrict__ Bm,
                                                        float* __restrict__ C,
                                                        int Nc,
                                                        const float* __restrict__ bias,
                                                        int relu) {
  __shared__ __align__(16) _Float16 As[16 * KD];
  __shared__ __align__(16) _Float16 Bs[64 * KD];
  int tid = threadIdx.x, lane = tid & 31, w = tid >> 5;
  int row0 = blockIdx.x * 16;
  int col0 = blockIdx.y * 64 + w * 16;

  { // stage A (16 contiguous rows): async b128, one chunk per thread per pass
    const uint4* src = (const uint4*)(A + (size_t)row0 * KD);
    uint4* dst = (uint4*)As;
#pragma unroll
    for (int i = 0; i < KD / 64; ++i)
      copy16_to_lds((const _Float16*)(src + tid + i * 128), (_Float16*)(dst + tid + i * 128));
  }
  { // stage this wave's 16 B columns, column-major [n][KD]
    int m = lane & 15, hi = lane >> 4;
    int klo = hi * (KD / 2), khi = klo + KD / 2;
    for (int k = klo; k < khi; ++k)
      Bs[(w * 16 + m) * KD + k] = Bm[(size_t)k * Nc + col0 + m];
  }
  wait_async_lds();
  __syncthreads();

  v8f acc = {};
#pragma unroll
  for (int kt = 0; kt < KD / 32; ++kt) {
    v16h af = load_frag16(As, KD, lane, kt);
    v16h bf = load_frag16(Bs + (size_t)w * 16 * KD, KD, lane, kt);
    acc = wmma_f16(af, bf, acc);
  }

  int m = lane & 15, mo = (lane >> 4) * 8;
#pragma unroll
  for (int v = 0; v < 8; ++v) {
    int r = row0 + mo + v, c = col0 + m;
    float val = acc[v];
    if (bias) val += bias[c];
    if (relu) val = fmaxf(val, 0.f);
    C[(size_t)r * Nc + c] = val;
  }
}

// ---------------- attention pooling ----------------
__global__ void gate_kernel(const float* __restrict__ h1, const float* __restrict__ Wg,
                            const float* __restrict__ bg, float* __restrict__ gate) {
  int t = blockIdx.x * blockDim.x + threadIdx.x;
  if (t >= BB * NN) return;
  float s = bg[0];
#pragma unroll 8
  for (int c = 0; c < D1; ++c) s += h1[(size_t)t * D1 + c] * Wg[c];
  gate[t] = fmaxf(s, 0.f);
}

__global__ __launch_bounds__(256) void softstat_kernel(const float* __restrict__ gate,
                                                       float* __restrict__ sstat) {
  __shared__ float red[256];
  int b = blockIdx.x, tid = threadIdx.x;
  const float* g = gate + (size_t)b * NN;
  float mx = -3.4e38f;
  for (int n = tid; n < NN; n += 256) mx = fmaxf(mx, g[n]);
  red[tid] = mx; __syncthreads();
  for (int s = 128; s > 0; s >>= 1) { if (tid < s) red[tid] = fmaxf(red[tid], red[tid + s]); __syncthreads(); }
  float bmax = red[0]; __syncthreads();
  float sm = 0.f;
  for (int n = tid; n < NN; n += 256) sm += expf(g[n] - bmax);
  red[tid] = sm; __syncthreads();
  for (int s = 128; s > 0; s >>= 1) { if (tid < s) red[tid] += red[tid + s]; __syncthreads(); }
  if (tid == 0) { sstat[b] = bmax; sstat[BB + b] = red[0]; }
}

__global__ void alpha_kernel(const float* __restrict__ gate, const float* __restrict__ sstat,
                             float* __restrict__ alpha) {
  int t = blockIdx.x * blockDim.x + threadIdx.x;
  if (t >= BB * NN) return;
  int b = t / NN;
  alpha[t] = expf(gate[t] - sstat[b]) / sstat[BB + b];
}

__global__ void pool_kernel(const float* __restrict__ alpha, const float* __restrict__ feat,
                            float* __restrict__ pooled) {
  int t = blockIdx.x * blockDim.x + threadIdx.x;
  if (t >= BB * DF) return;
  int b = t / DF, o = t % DF;
  float acc = 0.f;
  for (int n = 0; n < NN; ++n) {
    __builtin_prefetch(&feat[((size_t)b * NN + n + 8) * DF + o], 0, 1);
    acc += alpha[b * NN + n] * feat[((size_t)b * NN + n) * DF + o];
  }
  pooled[t] = acc;
}

__global__ void final_kernel(const float* __restrict__ pooled, const float* __restrict__ Wl,
                             const float* __restrict__ bl, float* __restrict__ out) {
  int t = blockIdx.x * blockDim.x + threadIdx.x;
  if (t >= BB * DOUT) return;
  int b = t / DOUT, o = t % DOUT;
  float acc = bl[o];
#pragma unroll 8
  for (int f = 0; f < DF; ++f) acc += pooled[b * DF + f] * Wl[f * DOUT + o];
  out[t] = acc;
}

// ---------------- host launcher ----------------
extern "C" void kernel_launch(void* const* d_in, const int* in_sizes, int n_in,
                              void* d_out, int out_size, void* d_ws, size_t ws_size,
                              hipStream_t stream) {
  const float* x   = (const float*)d_in[0];
  const float* Wt0 = (const float*)d_in[1];
  const float* bt0 = (const float*)d_in[2];
  const float* Wp0 = (const float*)d_in[3];
  const float* bp0 = (const float*)d_in[4];
  const float* g0  = (const float*)d_in[5];
  const float* be0 = (const float*)d_in[6];
  const float* Wt1 = (const float*)d_in[7];
  const float* bt1 = (const float*)d_in[8];
  const float* Wp1 = (const float*)d_in[9];
  const float* bp1 = (const float*)d_in[10];
  const float* g1  = (const float*)d_in[11];
  const float* be1 = (const float*)d_in[12];
  const float* Wg  = (const float*)d_in[13];
  const float* bg  = (const float*)d_in[14];
  const float* Wf  = (const float*)d_in[15];
  const float* bf  = (const float*)d_in[16];
  const float* Wl  = (const float*)d_in[17];
  const float* bl  = (const float*)d_in[18];
  float* out = (float*)d_out;

  // ---- workspace carve-up (all offsets 256B aligned) ----
  char* w = (char*)d_ws;
  size_t o = 0;
  auto alloc = [&](size_t bytes) { void* p = w + o; o = (o + bytes + 255) & ~(size_t)255; return p; };

  int*       idx0   = (int*)alloc((size_t)BB * NN * KNN * 4);
  float*     P0     = (float*)alloc((size_t)BB * NN * D0 * 4);
  float*     Q0     = (float*)alloc((size_t)BB * NN * D0 * 4);
  float*     hmax0  = (float*)alloc((size_t)BB * NN * D0 * 4);
  float*     stats0 = (float*)alloc(2 * D0 * 4);
  float*     h0     = (float*)alloc((size_t)BB * NN * D0 * 4);
  _Float16*  h0h    = (_Float16*)alloc((size_t)BB * NN * D0 * 2);
  float*     d2     = (float*)alloc((size_t)BB * NN * 4);
  int*       idx1   = (int*)alloc((size_t)BB * NN * KNN * 4);
  _Float16*  Wt1h   = (_Float16*)alloc((size_t)D0 * D1 * 2);
  _Float16*  Wdh    = (_Float16*)alloc((size_t)D0 * D1 * 2);
  float*     P1     = (float*)alloc((size_t)BB * NN * D1 * 4);
  float*     Q1     = (float*)alloc((size_t)BB * NN * D1 * 4);
  float*     hmax1  = (float*)alloc((size_t)BB * NN * D1 * 4);
  float*     stats1 = (float*)alloc(2 * D1 * 4);
  float*     h1     = (float*)alloc((size_t)BB * NN * D1 * 4);
  _Float16*  h1h    = (_Float16*)alloc((size_t)BB * NN * D1 * 2);
  _Float16*  Wfh    = (_Float16*)alloc((size_t)D1 * DF * 2);
  float*     gate   = (float*)alloc((size_t)BB * NN * 4);
  float*     sstat  = (float*)alloc(2 * BB * 4);
  float*     alphaw = (float*)alloc((size_t)BB * NN * 4);
  float*     feat   = (float*)alloc((size_t)BB * NN * DF * 4);
  float*     pooled = (float*)alloc((size_t)BB * DF * 4);
  (void)ws_size; (void)n_in; (void)in_sizes; (void)out_size;

  const int TPB = 256;

  // 0) zero BN accumulators (ws is poisoned; must be zeroed every call)
  hipLaunchKernelGGL(zero_stats_kernel, dim3(3), dim3(TPB), 0, stream, stats0, stats1);

  // 1) knn on raw points (D=3)
  hipLaunchKernelGGL(knn0_kernel, dim3((BB * NN + TPB - 1) / TPB), dim3(TPB), 0, stream, x, idx0);

  // 2) node projections for EdgeConv0 (m_ij = P_i + Q_j + b)
  hipLaunchKernelGGL(proj0_kernel, dim3((BB * NN * D0 + TPB - 1) / TPB), dim3(TPB), 0, stream,
                     x, Wt0, Wp0, P0, Q0);

  // 3) EdgeConv0 gather + max + BN stats; then BN(post-max)+leaky
  hipLaunchKernelGGL((edge_reduce_kernel<D0, 16>), dim3(BB * NN / 16), dim3(D0), 0, stream,
                     P0, Q0, bt0, bp0, idx0, hmax0, stats0);
  hipLaunchKernelGGL((bn_leaky_kernel<D0>), dim3((BB * NN * D0 + TPB - 1) / TPB), dim3(TPB), 0, stream,
                     hmax0, stats0, g0, be0, h0, h0h);

  // 4) knn on 64-d features: WMMA-fused distance + top-20 (async double-buffered)
  hipLaunchKernelGGL(d2_kernel, dim3((BB * NN + TPB - 1) / TPB), dim3(TPB), 0, stream, h0, d2);
  hipLaunchKernelGGL(knn1_kernel, dim3(BB * (NN / 32)), dim3(32), 0, stream, h0h, d2, idx1);

  // 5) EdgeConv1 projections via WMMA GEMM (Wt1 and Wp1-Wt1 in f16)
  hipLaunchKernelGGL(cvt_f16_kernel, dim3((D0 * D1 + TPB - 1) / TPB), dim3(TPB), 0, stream,
                     Wt1, Wt1h, D0 * D1);
  hipLaunchKernelGGL(cvt_diff_f16_kernel, dim3((D0 * D1 + TPB - 1) / TPB), dim3(TPB), 0, stream,
                     Wp1, Wt1, Wdh, D0 * D1);
  hipLaunchKernelGGL((gemm_wmma_kernel<D0>), dim3(BB * NN / 16, D1 / 64), dim3(128), 0, stream,
                     h0h, Wt1h, P1, D1, (const float*)nullptr, 0);
  hipLaunchKernelGGL((gemm_wmma_kernel<D0>), dim3(BB * NN / 16, D1 / 64), dim3(128), 0, stream,
                     h0h, Wdh, Q1, D1, (const float*)nullptr, 0);

  // 6) EdgeConv1 gather + max + BN stats; BN + leaky
  hipLaunchKernelGGL((edge_reduce_kernel<D1, 16>), dim3(BB * NN / 16), dim3(D1), 0, stream,
                     P1, Q1, bt1, bp1, idx1, hmax1, stats1);
  hipLaunchKernelGGL((bn_leaky_kernel<D1>), dim3((BB * NN * D1 + TPB - 1) / TPB), dim3(TPB), 0, stream,
                     hmax1, stats1, g1, be1, h1, h1h);

  // 7) attention pooling: gate, softmax stats, alpha, feat GEMM (WMMA, relu+bias)
  hipLaunchKernelGGL(gate_kernel, dim3((BB * NN + TPB - 1) / TPB), dim3(TPB), 0, stream,
                     h1, Wg, bg, gate);
  hipLaunchKernelGGL(softstat_kernel, dim3(BB), dim3(256), 0, stream, gate, sstat);
  hipLaunchKernelGGL(alpha_kernel, dim3((BB * NN + TPB - 1) / TPB), dim3(TPB), 0, stream,
                     gate, sstat, alphaw);
  hipLaunchKernelGGL(cvt_f16_kernel, dim3((D1 * DF + TPB - 1) / TPB), dim3(TPB), 0, stream,
                     Wf, Wfh, D1 * DF);
  hipLaunchKernelGGL((gemm_wmma_kernel<D1>), dim3(BB * NN / 16, DF / 64), dim3(128), 0, stream,
                     h1h, Wfh, feat, DF, bf, 1);
  hipLaunchKernelGGL(pool_kernel, dim3((BB * DF + TPB - 1) / TPB), dim3(TPB), 0, stream,
                     alphaw, feat, pooled);

  // 8) final linear -> d_out [B, 256]
  hipLaunchKernelGGL(final_kernel, dim3((BB * DOUT + TPB - 1) / TPB), dim3(TPB), 0, stream,
                     pooled, Wl, bl, out);
}
